// MultiHeadAttention_33320356282763
// MI455X (gfx1250) — compile-verified, matrix operand order checked by reading, not run on hardware
//
#include <hip/hip_runtime.h>

typedef unsigned short u16;
typedef unsigned int   u32;
typedef __attribute__((ext_vector_type(16))) __bf16 v16bf;
typedef __attribute__((ext_vector_type(8)))  float  v8f;
typedef __attribute__((ext_vector_type(4)))  u32    u32x4;
typedef __attribute__((ext_vector_type(2)))  u32    u32x2;
typedef __attribute__((ext_vector_type(4)))  float  f32x4;

#define E_DIM 1024
#define HEADS 16
#define HDIM  64
#define SEQ   512
#define MROWS (2 * 16 * SEQ)   // 16384

__device__ __forceinline__ u16 f2bf(float f) {
  u32 u = __float_as_uint(f);
  u += 0x7FFFu + ((u >> 16) & 1u);
  return (u16)(u >> 16);
}

// Build a 16-bit-type WMMA fragment (A row-slice or B^T n-slice) from LDS.
// CDNA5 layout: lane<16 holds k = base+{0..7,16..23}, lane>=16 holds
// k = base+{8..15,24..31}. Caller passes p = row_ptr + kb, kb = (lane>>4)*8.
__device__ __forceinline__ v16bf ldfrag(const u16* p) {
  union { u32x4 q[2]; v16bf v; } un;
  un.q[0] = *(const u32x4*)p;
  un.q[1] = *(const u32x4*)(p + 16);
  return un.v;
}

__device__ __forceinline__ v8f bwmma(v16bf a, v16bf b, v8f c) {
  return __builtin_amdgcn_wmma_f32_16x16x32_bf16(false, a, false, b, (short)0, c,
                                                 false, false);
}

// ---------------------------------------------------------------- convert ---
__global__ void cvt_bf16_kernel(const float* __restrict__ in,
                                u16* __restrict__ out, int n4) {
  int i = blockIdx.x * blockDim.x + threadIdx.x;
  if (i >= n4) return;
  f32x4 f = *(const f32x4*)(in + (size_t)i * 4);
  u32x2 r;
  r.x = (u32)f2bf(f.x) | ((u32)f2bf(f.y) << 16);
  r.y = (u32)f2bf(f.z) | ((u32)f2bf(f.w) << 16);
  *(u32x2*)(out + (size_t)i * 4) = r;
}

// fp32 [R][C] -> bf16 [C][R] tiled transpose-convert (one-off per weight)
__global__ __launch_bounds__(256) void cvt_t_kernel(const float* __restrict__ in,
                                                    u16* __restrict__ out,
                                                    int R, int C) {
  __shared__ float t[32][33];
  const int tid = threadIdx.x;
  const int tx = tid & 31, ty = tid >> 5;            // ty: 0..7
  const int br = blockIdx.x * 32, bc = blockIdx.y * 32;
#pragma unroll
  for (int j = 0; j < 4; ++j)
    t[ty + 8 * j][tx] = in[(size_t)(br + ty + 8 * j) * C + bc + tx];
  __syncthreads();
#pragma unroll
  for (int j = 0; j < 4; ++j)
    out[(size_t)(bc + ty + 8 * j) * R + br + tx] = f2bf(t[tx][ty + 8 * j]);
}

// ------------------------------------------------------------------- GEMM ---
// Y[M,N] = A[M,K](bf16) @ W^T-stored[N,K](bf16) + bias[N]
// MODE 0: bf16 out row-major   MODE 1: f32 out row-major
// MODE 2: bf16 out TRANSPOSED ([N][M], packed b128 stores)
#define BM 128
#define BN 128
#define BK 32
#define LDA 40
#define LDB 40

template <int MODE>
__global__ __launch_bounds__(256) void gemm_bias_kernel(
    const u16* __restrict__ A, const u16* __restrict__ WT,
    const float* __restrict__ bias,
    u16* __restrict__ outb, float* __restrict__ outf,
    int M, int N, int K)
{
  __shared__ __align__(16) u16 sA[BM * LDA];
  __shared__ __align__(16) u16 sB[BN * LDB];         // [n][k], pre-transposed W

  const int tid  = threadIdx.x;
  const int wave = tid >> 5, lane = tid & 31;
  const int wm = wave & 3, wn = wave >> 2;           // wave tile 32(m) x 64(n)
  const int bm = blockIdx.x * BM, bn = blockIdx.y * BN;
  const int l15 = lane & 15, kb = (lane >> 4) * 8, half = lane >> 4;

  // staging slots: 512 b128 each for A and B, 2 per thread
  const int arow = (tid * 2) >> 2, ac8 = ((tid * 2) & 3) * 8;
  const u16* gA = A  + (size_t)(bm + arow) * K + ac8;
  const u16* gB = WT + (size_t)(bn + arow) * K + ac8;
  u16* lA = &sA[arow * LDA + ac8];
  u16* lB = &sB[arow * LDA + ac8];

  v8f acc[2][4] = {};
  u32x4 ra[2], rb[2];
#pragma unroll
  for (int j = 0; j < 2; ++j) {                      // preload K-step 0
    ra[j] = *(const u32x4*)(gA + j * 8);
    rb[j] = *(const u32x4*)(gB + j * 8);
  }

  for (int kk = 0; kk < K; kk += BK) {
#pragma unroll
    for (int j = 0; j < 2; ++j) {                    // stage current tile
      *(u32x4*)(lA + j * 8) = ra[j];
      *(u32x4*)(lB + j * 8) = rb[j];
    }
    __syncthreads();
    if (kk + BK < K) {                               // prefetch next tile
#pragma unroll
      for (int j = 0; j < 2; ++j) {
        ra[j] = *(const u32x4*)(gA + kk + BK + j * 8);
        rb[j] = *(const u32x4*)(gB + kk + BK + j * 8);
      }
    }
    v16bf af[2], bf[4];
#pragma unroll
    for (int mi = 0; mi < 2; ++mi)
      af[mi] = ldfrag(&sA[(wm * 32 + mi * 16 + l15) * LDA + kb]);
#pragma unroll
    for (int ni = 0; ni < 4; ++ni)
      bf[ni] = ldfrag(&sB[(wn * 64 + ni * 16 + l15) * LDB + kb]);
#pragma unroll
    for (int mi = 0; mi < 2; ++mi)
#pragma unroll
      for (int ni = 0; ni < 4; ++ni)
        acc[mi][ni] = bwmma(af[mi], bf[ni], acc[mi][ni]);
    __syncthreads();
  }

#pragma unroll
  for (int mi = 0; mi < 2; ++mi) {
#pragma unroll
    for (int ni = 0; ni < 4; ++ni) {
      const int gn = bn + wn * 64 + ni * 16 + l15;
      const float bv = bias[gn];
      const float* ap = (const float*)&acc[mi][ni];
      const int gmb = bm + wm * 32 + mi * 16 + 8 * half;
      if (MODE == 2) {
        // transposed out: lane's 8 rows are contiguous -> one b128 store
        u32x4 r;
        r.x = (u32)f2bf(ap[0] + bv) | ((u32)f2bf(ap[1] + bv) << 16);
        r.y = (u32)f2bf(ap[2] + bv) | ((u32)f2bf(ap[3] + bv) << 16);
        r.z = (u32)f2bf(ap[4] + bv) | ((u32)f2bf(ap[5] + bv) << 16);
        r.w = (u32)f2bf(ap[6] + bv) | ((u32)f2bf(ap[7] + bv) << 16);
        *(u32x4*)(outb + (size_t)gn * M + gmb) = r;
      } else {
#pragma unroll
        for (int i = 0; i < 8; ++i) {
          const float v = ap[i] + bv;
          if (MODE == 1) outf[(size_t)(gmb + i) * N + gn] = v;
          else           outb[(size_t)(gmb + i) * N + gn] = f2bf(v);
        }
      }
    }
  }
}

// -------------------------------------------------------- flash attention ---
// 256 threads = 8 waves per block; block = 128 queries x full key sweep.
#define LDQ 72

__global__ __launch_bounds__(256) void attn_kernel(
    const u16* __restrict__ Q, const u16* __restrict__ Kp,
    const u16* __restrict__ Vt,                      // transposed: [E][M]
    u16* __restrict__ ctx)
{
  __shared__ __align__(16) u16 sQ [128 * LDQ];
  __shared__ __align__(16) u16 sK [64 * LDQ];
  __shared__ __align__(16) u16 sVt[64 * LDQ];        // [d][key]
  __shared__ __align__(16) u16 sP [8][16 * LDQ];

  const int tid  = threadIdx.x;
  const int wave = tid >> 5, lane = tid & 31;
  const int l15 = lane & 15, kb = (lane >> 4) * 8, half = lane >> 4;

  const int brh = blockIdx.x;
  const int h  = brh & (HEADS - 1);
  const int br = brh >> 4;
  const size_t rowbase = (size_t)br * SEQ;
  const int colbase = h * HDIM;
  const int qbase = blockIdx.y * 128;

  const int srow = (tid * 2) >> 3, sd8 = ((tid * 2) & 7) * 8;  // K/V slot base

  // stage Q tile 128x64 (1024 b128 slots / 256 threads)
#pragma unroll
  for (int j = 0; j < 4; ++j) {
    int s = tid * 4 + j;
    int row = s >> 3, d8 = (s & 7) * 8;
    u32x4 v = *(const u32x4*)(Q + (rowbase + qbase + row) * E_DIM + colbase + d8);
    *(u32x4*)&sQ[row * LDQ + d8] = v;
  }
  __syncthreads();
  const v16bf aq0 = ldfrag(&sQ[(wave * 16 + l15) * LDQ + kb]);
  const v16bf aq1 = ldfrag(&sQ[(wave * 16 + l15) * LDQ + 32 + kb]);

  float mrow[8], lrow[8];
#pragma unroll
  for (int i = 0; i < 8; ++i) { mrow[i] = -__builtin_inff(); lrow[i] = 0.f; }
  v8f oacc[4] = {};

  for (int kt = 0; kt < SEQ / 64; ++kt) {
    const int kbase = kt * 64;
#pragma unroll
    for (int j = 0; j < 2; ++j) {                    // K tile [key][d]
      int s = tid * 2 + j;
      int row = s >> 3, d8 = (s & 7) * 8;
      u32x4 v = *(const u32x4*)(Kp + (rowbase + kbase + row) * E_DIM + colbase + d8);
      *(u32x4*)&sK[row * LDQ + d8] = v;
    }
#pragma unroll
    for (int j = 0; j < 2; ++j) {                    // V^T tile [d][key], direct
      int s = tid * 2 + j;
      int drow = s >> 3, key8 = (s & 7) * 8;
      u32x4 v = *(const u32x4*)(Vt + (size_t)(colbase + drow) * MROWS +
                                rowbase + kbase + key8);
      *(u32x4*)&sVt[drow * LDQ + key8] = v;
    }
    if (kt + 1 < SEQ / 64) {                         // prefetch next tiles (WGP)
      __builtin_prefetch(Kp + (rowbase + kbase + 64 + srow) * E_DIM + colbase + sd8, 0, 3);
      __builtin_prefetch(Vt + (size_t)(colbase + srow) * MROWS + rowbase + kbase + 64 + sd8, 0, 3);
    }
    __syncthreads();

    // S = Q K^T  (wave: 16 query rows x 64 keys)
    v8f sacc[4] = {};
#pragma unroll
    for (int ni = 0; ni < 4; ++ni) {
      v16bf bk0 = ldfrag(&sK[(ni * 16 + l15) * LDQ + kb]);
      v16bf bk1 = ldfrag(&sK[(ni * 16 + l15) * LDQ + 32 + kb]);
      sacc[ni] = bwmma(aq0, bk0, sacc[ni]);
      sacc[ni] = bwmma(aq1, bk1, sacc[ni]);
    }

    // online softmax in C/D register layout: row (i + 8*half) spans one
    // 16-lane half -> xor-shuffle masks 1..8 reduce per row.
    float mnew[8], psum[8];
#pragma unroll
    for (int i = 0; i < 8; ++i) {
      float mx = -__builtin_inff();
#pragma unroll
      for (int ni = 0; ni < 4; ++ni)
        mx = fmaxf(mx, ((const float*)&sacc[ni])[i] * 0.125f);
      mx = fmaxf(mx, __shfl_xor(mx, 1, 32));
      mx = fmaxf(mx, __shfl_xor(mx, 2, 32));
      mx = fmaxf(mx, __shfl_xor(mx, 4, 32));
      mx = fmaxf(mx, __shfl_xor(mx, 8, 32));
      mnew[i] = fmaxf(mrow[i], mx);
      psum[i] = 0.f;
    }
#pragma unroll
    for (int ni = 0; ni < 4; ++ni) {
#pragma unroll
      for (int i = 0; i < 8; ++i) {
        float p = __expf(((const float*)&sacc[ni])[i] * 0.125f - mnew[i]);
        psum[i] += p;
        sP[wave][(i + 8 * half) * LDQ + ni * 16 + l15] = f2bf(p);
      }
    }
#pragma unroll
    for (int i = 0; i < 8; ++i) {
      float ps = psum[i];
      ps += __shfl_xor(ps, 1, 32);
      ps += __shfl_xor(ps, 2, 32);
      ps += __shfl_xor(ps, 4, 32);
      ps += __shfl_xor(ps, 8, 32);
      float alpha = __expf(mrow[i] - mnew[i]);
      lrow[i] = lrow[i] * alpha + ps;
      mrow[i] = mnew[i];
#pragma unroll
      for (int t = 0; t < 4; ++t)
        ((float*)&oacc[t])[i] *= alpha;
    }

    // O += P V
    v16bf ap0 = ldfrag(&sP[wave][l15 * LDQ + kb]);
    v16bf ap1 = ldfrag(&sP[wave][l15 * LDQ + 32 + kb]);
#pragma unroll
    for (int t = 0; t < 4; ++t) {
      v16bf bv0 = ldfrag(&sVt[(t * 16 + l15) * LDQ + kb]);
      v16bf bv1 = ldfrag(&sVt[(t * 16 + l15) * LDQ + 32 + kb]);
      oacc[t] = bwmma(ap0, bv0, oacc[t]);
      oacc[t] = bwmma(ap1, bv1, oacc[t]);
    }
    __syncthreads();
  }

#pragma unroll
  for (int t = 0; t < 4; ++t) {
#pragma unroll
    for (int i = 0; i < 8; ++i) {
      float v = ((const float*)&oacc[t])[i] / lrow[i];
      size_t qr = rowbase + qbase + wave * 16 + i + 8 * half;
      ctx[qr * E_DIM + colbase + t * 16 + l15] = f2bf(v);
    }
  }
}

// ------------------------------------------------------------------ launch --
extern "C" void kernel_launch(void* const* d_in, const int* in_sizes, int n_in,
                              void* d_out, int out_size, void* d_ws, size_t ws_size,
                              hipStream_t stream) {
  const float* x  = (const float*)d_in[0];
  const float* wq = (const float*)d_in[1];
  const float* bq = (const float*)d_in[2];
  const float* wk = (const float*)d_in[3];
  const float* bk = (const float*)d_in[4];
  const float* wv = (const float*)d_in[5];
  const float* bv = (const float*)d_in[6];
  const float* wo = (const float*)d_in[7];
  const float* bo = (const float*)d_in[8];
  float* out = (float*)d_out;

  const int M = MROWS;
  const size_t ME = (size_t)M * E_DIM;
  const size_t EE = (size_t)E_DIM * E_DIM;

  u16* xb  = (u16*)d_ws;                 // reused as ctx after attention
  u16* wqT = xb  + ME;
  u16* wkT = wqT + EE;
  u16* wvT = wkT + EE;
  u16* woT = wvT + EE;
  u16* Qb  = woT + EE;
  u16* Kb  = Qb  + ME;
  u16* Vt  = Kb  + ME;                   // V stored transposed [E][M]
  u16* ctx = xb;

  // 1) fp32 -> bf16 (x row-major; weights transpose-converted to [N][K])
  cvt_bf16_kernel<<<(int)(ME / 4 / 256), 256, 0, stream>>>(x, xb, (int)(ME / 4));
  dim3 gt(E_DIM / 32, E_DIM / 32);
  cvt_t_kernel<<<gt, 256, 0, stream>>>(wq, wqT, E_DIM, E_DIM);
  cvt_t_kernel<<<gt, 256, 0, stream>>>(wk, wkT, E_DIM, E_DIM);
  cvt_t_kernel<<<gt, 256, 0, stream>>>(wv, wvT, E_DIM, E_DIM);
  cvt_t_kernel<<<gt, 256, 0, stream>>>(wo, woT, E_DIM, E_DIM);

  // 2) Q/K/V projections (V written transposed for the attention B-operand)
  dim3 gg(M / BM, E_DIM / BN);
  gemm_bias_kernel<0><<<gg, 256, 0, stream>>>(xb, wqT, bq, Qb, nullptr, M, E_DIM, E_DIM);
  gemm_bias_kernel<0><<<gg, 256, 0, stream>>>(xb, wkT, bk, Kb, nullptr, M, E_DIM, E_DIM);
  gemm_bias_kernel<2><<<gg, 256, 0, stream>>>(xb, wvT, bv, Vt, nullptr, M, E_DIM, E_DIM);

  // 3) flash attention: grid = (b*r*h, query tiles of 128)
  attn_kernel<<<dim3(2 * 16 * HEADS, SEQ / 128), 256, 0, stream>>>(Qb, Kb, Vt, ctx);

  // 4) output projection, fp32 out
  gemm_bias_kernel<1><<<gg, 256, 0, stream>>>(ctx, woT, bo, nullptr, out, M, E_DIM, E_DIM);
}
